// SetMatchingModel_515396076027
// MI455X (gfx1250) — matrix operand order repaired and checked by reading, not compiled, not run on hardware
//
#include <hip/hip_runtime.h>
#include <math.h>

// ---------------------------------------------------------------------------
// CDNA5 (gfx1250) SetMatchingModel implementation.
// All GEMMs use v_wmma_f32_16x16x32_bf16 (wave32, f32 accum),
// register-blocked NB output tiles per wave for independent wmma chains.
// ---------------------------------------------------------------------------

typedef __attribute__((ext_vector_type(16))) __bf16 v16bf;
typedef __attribute__((ext_vector_type(8)))  float  v8f;

#define SETS   48
#define NITEM  80
#define NPAD   96          // 80 padded to multiple of 32 for K-dim of A*V
#define DMODEL 128
#define HSZ    128
#define WAVES  8           // 256 threads = 8 wave32

// bf16 weight pool element offsets inside workspace
enum : int {
  OW1    = 0,
  OW2    = OW1    + 512*512,
  OW3    = OW2    + 512*256,
  OWQ_S  = OW3    + 256*128,
  OWK_S  = OWQ_S  + 128*256,
  OWV_S  = OWK_S  + 128*256,
  OWH_S  = OWV_S  + 128*256,
  OWFC_E = OWH_S  + 256*128,
  OWQ_C  = OWFC_E + 128*128,
  OWK_C  = OWQ_C  + 128*256,
  OWV_C  = OWK_C  + 128*256,
  OWH_C  = OWV_C  + 128*256,
  OWFC_D = OWH_C  + 256*128,
  OWCS   = OWFC_D + 128*128,
  WT_ELEMS = OWCS + 128*256
};

__device__ __forceinline__ float gelu_exact(float x) {
  return 0.5f * x * (1.0f + erff(x * 0.70710678118654752f));
}

// A fragment: 16x32 bf16, row-major source, CDNA5 layout:
// lanes 0-15: M=lane, K = k0+{0..7, 16..23}; lanes 16-31: K = k0+{8..15, 24..31}
__device__ __forceinline__ v16bf load_a_frag(const __bf16* A, int lda, int m0, int k0, int lane) {
  int m  = m0 + (lane & 15);
  int kb = k0 + ((lane & 16) ? 8 : 0);
  const __bf16* p = A + m * lda + kb;
  v16bf f;
#pragma unroll
  for (int e = 0; e < 8; ++e) f[e] = p[e];
#pragma unroll
  for (int e = 0; e < 8; ++e) f[8 + e] = p[16 + e];
  return f;
}

// B fragment: 32x16 bf16 (K x N), row-major source; N = lane&15, K same order as A frag
__device__ __forceinline__ v16bf load_b_frag(const __bf16* B, int ldb, int k0, int n0, int lane) {
  const __bf16* p = B + (k0 + ((lane & 16) ? 8 : 0)) * ldb + n0 + (lane & 15);
  v16bf f;
#pragma unroll
  for (int e = 0; e < 8; ++e) f[e] = p[e * ldb];
#pragma unroll
  for (int e = 0; e < 8; ++e) f[8 + e] = p[(16 + e) * ldb];
  return f;
}

// B = X^T where X is row-major [N][Kdim] (used by score stage)
__device__ __forceinline__ v16bf load_bT_frag(const __bf16* X, int ldx, int k0, int n0, int lane) {
  const __bf16* p = X + (n0 + (lane & 15)) * ldx + k0 + ((lane & 16) ? 8 : 0);
  v16bf f;
#pragma unroll
  for (int e = 0; e < 8; ++e) f[e] = p[e];
#pragma unroll
  for (int e = 0; e < 8; ++e) f[8 + e] = p[16 + e];
  return f;
}

#define ST_F32      0
#define ST_BF16     1
#define ST_BF16_HT  2   // per-head transposed store: dst[col*80 + row] (col = h*128+d)

// Block-cooperative GEMM: C[M,N] = epi(scale * A[M,K] @ B[K,N]) (+ Res).
// M,N multiples of 16; K multiple of 32; (N/16) % NB == 0.
// Each wave owns NB adjacent 16x16 output tiles: one A-fragment load feeds NB
// independent v_wmma chains per k-step (hides WMMA RAW latency, cuts A traffic).
template<int MODE, int ACT, bool HASRES, bool PFB, int NB>
__device__ __forceinline__ void gemm_block(const __bf16* __restrict__ A, int lda,
                                           const __bf16* __restrict__ B, int ldb,
                                           int M, int N, int K,
                                           float* Cf, __bf16* Cb, int ldc,
                                           const float* Res, int ldres,
                                           float scale, int wave, int lane) {
  const int mt = M >> 4, ntb = (N >> 4) / NB, kt = K >> 5;
  for (int t = wave; t < mt * ntb; t += WAVES) {
    const int m0 = (t / ntb) << 4;
    const int n0 = (t % ntb) * (16 * NB);
    v8f acc[NB];
#pragma unroll
    for (int u = 0; u < NB; ++u) acc[u] = (v8f){0.f, 0.f, 0.f, 0.f, 0.f, 0.f, 0.f, 0.f};
    for (int kk = 0; kk < kt; ++kk) {
      if (PFB && (kk + 1) < kt) {
        __builtin_prefetch(B + (((kk + 1) << 5) + lane) * ldb + n0, 0, 1);
        if (NB == 4)
          __builtin_prefetch(B + (((kk + 1) << 5) + lane) * ldb + n0 + 64, 0, 1);
      }
      v16bf a = load_a_frag(A, lda, m0, kk << 5, lane);
#pragma unroll
      for (int u = 0; u < NB; ++u) {
        v16bf b = load_b_frag(B, ldb, kk << 5, n0 + u * 16, lane);
        acc[u] = __builtin_amdgcn_wmma_f32_16x16x32_bf16(false, a, false, b, (short)0,
                                                         acc[u], false, false);
      }
    }
    const int rb = m0 + ((lane & 16) ? 8 : 0);
#pragma unroll
    for (int u = 0; u < NB; ++u) {
      const int col = n0 + u * 16 + (lane & 15);
#pragma unroll
      for (int r = 0; r < 8; ++r) {
        const int row = rb + r;
        float v = scale * acc[u][r];
        if (ACT) v = gelu_exact(v);
        if (HASRES) v += Res[row * ldres + col];
        if (MODE == ST_F32)       Cf[row * ldc + col] = v;
        else if (MODE == ST_BF16) Cb[row * ldc + col] = (__bf16)v;
        else                      Cb[col * 80 + row]  = (__bf16)v;
      }
    }
  }
}

// per-item layer norm over D=128, zero rows whose sum == 0 (reference semantics)
__device__ __forceinline__ void item_norm_block(const float* X, __bf16* Out, int wave, int lane) {
  for (int r = wave; r < NITEM; r += WAVES) {
    const float* xr = X + r * DMODEL;
    float s = 0.f, s2 = 0.f;
#pragma unroll
    for (int c = lane; c < DMODEL; c += 32) { float v = xr[c]; s += v; s2 += v * v; }
#pragma unroll
    for (int off = 16; off > 0; off >>= 1) {
      s  += __shfl_xor(s,  off, 32);
      s2 += __shfl_xor(s2, off, 32);
    }
    const float mean = s * (1.f / DMODEL);
    const float var  = fmaxf(s2 * (1.f / DMODEL) - mean * mean, 0.f);
    const float inv  = 1.f / (sqrtf(var) + 1e-3f);
    const bool keep  = (s != 0.f);
#pragma unroll
    for (int c = lane; c < DMODEL; c += 32) {
      float v = keep ? (xr[c] - mean) * inv : 0.f;
      Out[r * DMODEL + c] = (__bf16)v;
    }
  }
}

// masked softmax rows: mask where score == 0 (reference semantics); pad cols 80..95 with 0
__device__ __forceinline__ void softmax_block(const float* Sb, __bf16* Ab, int wave, int lane) {
  for (int r = wave; r < NITEM; r += WAVES) {
    const float* sr = Sb + r * NPAD;
    const bool has3 = (lane < 16);
    float v0 = sr[lane];
    float v1 = sr[lane + 32];
    float v2 = has3 ? sr[lane + 64] : -3.4e38f;
    float mx = fmaxf(fmaxf(v0, v1), v2);
#pragma unroll
    for (int off = 16; off > 0; off >>= 1) mx = fmaxf(mx, __shfl_xor(mx, off, 32));
    float e0 = (v0 != 0.f) ? expf(v0 - mx) : 0.f;
    float e1 = (v1 != 0.f) ? expf(v1 - mx) : 0.f;
    float e2 = (has3 && v2 != 0.f) ? expf(v2 - mx) : 0.f;
    float sum = e0 + e1 + e2;
#pragma unroll
    for (int off = 16; off > 0; off >>= 1) sum += __shfl_xor(sum, off, 32);
    const float inv = 1.f / (sum + 1e-10f);
    Ab[r * NPAD + lane]      = (__bf16)(e0 * inv);
    Ab[r * NPAD + lane + 32] = (__bf16)(e1 * inv);
    if (has3) {
      Ab[r * NPAD + lane + 64] = (__bf16)(e2 * inv);
      Ab[r * NPAD + 80 + lane] = (__bf16)0.0f;  // zero pad cols 80..95
    }
  }
}

// relu-sum of (Xa_h @ Xb_h^T)/sqrt(HS): one partial per thread
__device__ __forceinline__ float score_partial(const __bf16* Xa, const __bf16* Xb,
                                               int hh, int wave, int lane) {
  float local = 0.f;
  const float scale = 0.088388347648318447f;  // 1/sqrt(128)
  for (int t = wave; t < 25; t += WAVES) {
    const int m0 = (t / 5) << 4, n0 = (t % 5) << 4;
    v8f acc = {0.f, 0.f, 0.f, 0.f, 0.f, 0.f, 0.f, 0.f};
#pragma unroll
    for (int kk = 0; kk < 4; ++kk) {
      v16bf a = load_a_frag(Xa + hh * HSZ, 2 * HSZ, m0, kk << 5, lane);
      v16bf b = load_bT_frag(Xb + hh * HSZ, 2 * HSZ, kk << 5, n0, lane);
      acc = __builtin_amdgcn_wmma_f32_16x16x32_bf16(false, a, false, b, (short)0, acc,
                                                    false, false);
    }
#pragma unroll
    for (int r = 0; r < 8; ++r) local += fmaxf(scale * acc[r], 0.f);
  }
  return local;
}

// ---------------------------------------------------------------------------
__global__ void cvt_kernel(const float* __restrict__ src, __bf16* __restrict__ dst, int n) {
  int i = blockIdx.x * 256 + threadIdx.x;
  if (i < n) dst[i] = (__bf16)src[i];
}

// MLP: x[3840,512] -> gelu GEMM chain -> h[3840,128]. 32 rows per block.
__global__ __launch_bounds__(256) void mlp_kernel(const float* __restrict__ x,
                                                  const __bf16* __restrict__ WB,
                                                  float* __restrict__ h) {
  extern __shared__ char smem[];
  float*  fbuf = (float*)(smem);               // 32*512 f32
  __bf16* inb  = (__bf16*)(smem + 32*512*4);   // 32*512 bf16
  const int tid = threadIdx.x, wave = tid >> 5, lane = tid & 31;
  const int base = blockIdx.x * 32;

  for (int i = tid; i < 32 * 512; i += 256) inb[i] = (__bf16)x[base * 512 + i];
  __syncthreads();
  gemm_block<ST_F32, 1, false, true, 4>(inb, 512, WB + OW1, 512, 32, 512, 512,
                                        fbuf, nullptr, 512, nullptr, 0, 1.f, wave, lane);
  __syncthreads();
  for (int i = tid; i < 32 * 512; i += 256) inb[i] = (__bf16)fbuf[i];
  __syncthreads();
  gemm_block<ST_F32, 1, false, true, 4>(inb, 512, WB + OW2, 256, 32, 256, 512,
                                        fbuf, nullptr, 256, nullptr, 0, 1.f, wave, lane);
  __syncthreads();
  for (int i = tid; i < 32 * 256; i += 256) inb[i] = (__bf16)fbuf[i];
  __syncthreads();
  gemm_block<ST_F32, 1, false, true, 2>(inb, 256, WB + OW3, 128, 32, 128, 256,
                                        h + base * 128, nullptr, 128,
                                        nullptr, 0, 1.f, wave, lane);
}

// Encoder + decoder-QKV precompute, one set per block.
__global__ __launch_bounds__(256) void encoder_kernel(const float* __restrict__ hbuf,
                                                      const __bf16* __restrict__ WB,
                                                      float* __restrict__ ebuf,
                                                      __bf16* __restrict__ Qg,
                                                      __bf16* __restrict__ KTg,
                                                      __bf16* __restrict__ Vg) {
  extern __shared__ char smem[];
  float*  ef = (float*)(smem + 0);        // [80][128] f32
  float*  Sb = (float*)(smem + 40960);    // [80][96]  f32
  __bf16* zb = (__bf16*)(smem + 71680);   // [80][128] bf16
  __bf16* qb = (__bf16*)(smem + 92160);   // [80][256] bf16
  __bf16* kT = (__bf16*)(smem + 133120);  // [2][128][80] bf16
  __bf16* vb = (__bf16*)(smem + 174080);  // [96][256] bf16 (rows 80..95 zeroed)
  __bf16* Ab = (__bf16*)(smem + 223232);  // [80][96]  bf16
  __bf16* Ob = (__bf16*)(smem + 238592);  // [80][256] bf16
  const int tid = threadIdx.x, wave = tid >> 5, lane = tid & 31;
  const int s = blockIdx.x;
  const float ishs = 0.088388347648318447f;

  const float* hs = hbuf + s * NITEM * DMODEL;
  for (int i = tid; i < NITEM * DMODEL; i += 256) ef[i] = hs[i];
  for (int i = tid; i < 16 * 256; i += 256) vb[NITEM * 256 + i] = (__bf16)0.0f;
  __syncthreads();

  item_norm_block(ef, zb, wave, lane);
  __syncthreads();
  gemm_block<ST_BF16, 0, false, true, 4>(zb, 128, WB + OWQ_S, 256, 80, 256, 128,
                                         nullptr, qb, 256, nullptr, 0, 1.f, wave, lane);
  gemm_block<ST_BF16_HT, 0, false, true, 4>(zb, 128, WB + OWK_S, 256, 80, 256, 128,
                                            nullptr, kT, 0, nullptr, 0, 1.f, wave, lane);
  gemm_block<ST_BF16, 0, false, true, 4>(zb, 128, WB + OWV_S, 256, 80, 256, 128,
                                         nullptr, vb, 256, nullptr, 0, 1.f, wave, lane);
  __syncthreads();

  for (int hh = 0; hh < 2; ++hh) {
    gemm_block<ST_F32, 0, false, false, 1>(qb + hh * HSZ, 256, kT + hh * HSZ * 80, 80,
                                           80, 80, 128, Sb, nullptr, NPAD,
                                           nullptr, 0, ishs, wave, lane);
    __syncthreads();
    softmax_block(Sb, Ab, wave, lane);
    __syncthreads();
    gemm_block<ST_BF16, 0, false, false, 2>(Ab, NPAD, vb + hh * HSZ, 256, 80, 128, NPAD,
                                            nullptr, Ob + hh * HSZ, 256,
                                            nullptr, 0, 1.f, wave, lane);
    __syncthreads();
  }

  gemm_block<ST_F32, 0, true, true, 2>(Ob, 256, WB + OWH_S, 128, 80, 128, 256,
                                       ef, nullptr, 128, ef, 128, 1.f, wave, lane);
  __syncthreads();
  item_norm_block(ef, zb, wave, lane);
  __syncthreads();
  gemm_block<ST_F32, 1, true, true, 2>(zb, 128, WB + OWFC_E, 128, 80, 128, 128,
                                       ef, nullptr, 128, ef, 128, 1.f, wave, lane);
  __syncthreads();

  float* es = ebuf + s * NITEM * DMODEL;
  for (int i = tid; i < NITEM * DMODEL; i += 256) es[i] = ef[i];
  item_norm_block(ef, zb, wave, lane);
  __syncthreads();

  gemm_block<ST_BF16, 0, false, true, 4>(zb, 128, WB + OWQ_C, 256, 80, 256, 128,
                                         nullptr, Qg + s * NITEM * 256, 256,
                                         nullptr, 0, 1.f, wave, lane);
  gemm_block<ST_BF16_HT, 0, false, true, 4>(zb, 128, WB + OWK_C, 256, 80, 256, 128,
                                            nullptr, KTg + s * 2 * HSZ * 80, 0,
                                            nullptr, 0, 1.f, wave, lane);
  gemm_block<ST_BF16, 0, false, true, 4>(zb, 128, WB + OWV_C, 256, 80, 256, 128,
                                         nullptr, Vg + s * NITEM * 256, 256,
                                         nullptr, 0, 1.f, wave, lane);
  // zero the 16-row tail after the last set so AV padded-K reads never hit NaN bits
  if (s == SETS - 1)
    for (int i = tid; i < 16 * 256; i += 256) Vg[SETS * NITEM * 256 + i] = (__bf16)0.0f;
}

// One unordered pair {i,j} per block: cross attention both directions + symmetric score.
__global__ __launch_bounds__(256) void pair_kernel(const float* __restrict__ ebuf,
                                                   const __bf16* __restrict__ WB,
                                                   const __bf16* __restrict__ Qg,
                                                   const __bf16* __restrict__ KTg,
                                                   const __bf16* __restrict__ Vg,
                                                   const float* __restrict__ xsz,
                                                   const float* __restrict__ wcs2,
                                                   float* __restrict__ out) {
  extern __shared__ char smem[];
  float*  ssum = (float*)(smem);           // [2]
  float*  Sb   = (float*)(smem + 16);      // [80][96]  f32
  float*  d4   = (float*)(smem + 30736);   // [80][128] f32
  __bf16* Ab   = (__bf16*)(smem + 71696);  // [80][96]
  __bf16* Ob   = (__bf16*)(smem + 87056);  // [80][256]
  __bf16* zb   = (__bf16*)(smem + 128016); // [80][128]
  __bf16* xb   = (__bf16*)(smem + 148496); // [80][128]
  __bf16* xh   = (__bf16*)(smem + 168976); // [2][80][256]
  const int tid = threadIdx.x, wave = tid >> 5, lane = tid & 31;
  const float ishs = 0.088388347648318447f;

  int p = blockIdx.x, i = 0;
  while (p >= (SETS - i)) { p -= (SETS - i); ++i; }
  const int j = i + p;

  if (tid < 2) ssum[tid] = 0.f;
  __syncthreads();

  for (int dir = 0; dir < 2; ++dir) {
    const int a = dir ? j : i;
    const int b = dir ? i : j;
    const __bf16* Qa  = Qg  + a * NITEM * 256;
    const __bf16* KTb = KTg + b * 2 * HSZ * 80;
    const __bf16* Vb  = Vg  + b * NITEM * 256;
    const float*  ea  = ebuf + a * NITEM * DMODEL;

    for (int hh = 0; hh < 2; ++hh) {
      gemm_block<ST_F32, 0, false, false, 1>(Qa + hh * HSZ, 256, KTb + hh * HSZ * 80, 80,
                                             80, 80, 128, Sb, nullptr, NPAD,
                                             nullptr, 0, ishs, wave, lane);
      __syncthreads();
      softmax_block(Sb, Ab, wave, lane);
      __syncthreads();
      gemm_block<ST_BF16, 0, false, false, 2>(Ab, NPAD, Vb + hh * HSZ, 256, 80, 128, NPAD,
                                              nullptr, Ob + hh * HSZ, 256,
                                              nullptr, 0, 1.f, wave, lane);
      __syncthreads();
    }
    gemm_block<ST_F32, 0, true, true, 2>(Ob, 256, WB + OWH_C, 128, 80, 128, 256,
                                         d4, nullptr, 128, ea, 128, 1.f, wave, lane);
    __syncthreads();
    item_norm_block(d4, zb, wave, lane);
    __syncthreads();
    gemm_block<ST_F32, 1, true, true, 2>(zb, 128, WB + OWFC_D, 128, 80, 128, 128,
                                         d4, nullptr, 128, d4, 128, 1.f, wave, lane);
    __syncthreads();
    for (int q = tid; q < NITEM * DMODEL; q += 256) xb[q] = (__bf16)d4[q];
    __syncthreads();
    gemm_block<ST_BF16, 0, false, true, 4>(xb, 128, WB + OWCS, 256, 80, 256, 128,
                                           nullptr, xh + dir * NITEM * 256, 256,
                                           nullptr, 0, 1.f, wave, lane);
    __syncthreads();
  }

  // symmetric relu score: same value for (i,j) and (j,i)
  float l0 = score_partial(xh, xh + NITEM * 256, 0, wave, lane);
  float l1 = score_partial(xh, xh + NITEM * 256, 1, wave, lane);
  atomicAdd(&ssum[0], l0);
  atomicAdd(&ssum[1], l1);
  __syncthreads();
  if (tid == 0) {
    const float norm = 1.f / (xsz[i] * xsz[j]);
    const float v = ssum[0] * norm * wcs2[0] + ssum[1] * norm * wcs2[1];
    out[i * SETS + j] = v;
    out[j * SETS + i] = v;
  }
}

// ---------------------------------------------------------------------------
extern "C" void kernel_launch(void* const* d_in, const int* in_sizes, int n_in,
                              void* d_out, int out_size, void* d_ws, size_t ws_size,
                              hipStream_t stream) {
  (void)in_sizes; (void)n_in; (void)out_size; (void)ws_size;
  const float* x    = (const float*)d_in[0];
  const float* xsz  = (const float*)d_in[1];
  const float* wcs2 = (const float*)d_in[16];

  char* ws = (char*)d_ws;
  __bf16* WB = (__bf16*)ws;
  size_t off = (size_t)WT_ELEMS * sizeof(__bf16);
  off = (off + 255) & ~(size_t)255;
  float* hbuf = (float*)(ws + off); off += (size_t)SETS * NITEM * DMODEL * sizeof(float);
  float* ebuf = (float*)(ws + off); off += (size_t)SETS * NITEM * DMODEL * sizeof(float);
  __bf16* Qg  = (__bf16*)(ws + off); off += (size_t)SETS * NITEM * 256 * sizeof(__bf16);
  __bf16* KTg = (__bf16*)(ws + off); off += (size_t)SETS * 2 * HSZ * 80 * sizeof(__bf16);
  __bf16* Vg  = (__bf16*)(ws + off); off += (size_t)(SETS * NITEM + 16) * 256 * sizeof(__bf16);

  static const struct { int idx; int woff; int n; } kSegs[14] = {
    { 2, OW1,   512*512}, { 3, OW2,   512*256}, { 4, OW3,   256*128},
    { 5, OWQ_S, 128*256}, { 6, OWK_S, 128*256}, { 7, OWV_S, 128*256},
    { 8, OWH_S, 256*128}, { 9, OWFC_E,128*128},
    {10, OWQ_C, 128*256}, {11, OWK_C, 128*256}, {12, OWV_C, 128*256},
    {13, OWH_C, 256*128}, {14, OWFC_D,128*128}, {15, OWCS,  128*256},
  };
  for (int s = 0; s < 14; ++s) {
    int n = kSegs[s].n;
    cvt_kernel<<<(n + 255) / 256, 256, 0, stream>>>((const float*)d_in[kSegs[s].idx],
                                                    WB + kSegs[s].woff, n);
  }

  mlp_kernel<<<(SETS * NITEM) / 32, 256, 32*512*4 + 32*512*2, stream>>>(x, WB, hbuf);
  encoder_kernel<<<SETS, 256, 279552, stream>>>(hbuf, WB, ebuf, Qg, KTg, Vg);
  pair_kernel<<<(SETS * (SETS + 1)) / 2, 256, 250896, stream>>>(ebuf, WB, Qg, KTg, Vg,
                                                                xsz, wcs2, (float*)d_out);
}
